// StockTransformer_24163486007575
// MI455X (gfx1250) — matrix operand order repaired and checked by reading, not compiled, not run on hardware
//
#include <hip/hip_runtime.h>
#include <hip/hip_bf16.h>

// ---------------------------------------------------------------------------
// StockTransformer top-k blocked attention for MI455X (gfx1250, wave32).
// GEMMs (sim, QKV proj, out proj) use v_wmma_f32_16x16x32_bf16.
// Workspace requirement: ~185 MB (dominated by B*3008*3008 fp32 sim matrix).
// ---------------------------------------------------------------------------

#define B_ 4
#define N_ 3000
#define D_ 128
#define H_ 4
#define HD_ 32
#define TOPK_ 40
#define NPAD 3008          // 188 * 16
#define QTILES 188
#define NEGF (-3.402823466e38f)

typedef __attribute__((ext_vector_type(16))) __bf16 v16bf;
typedef __attribute__((ext_vector_type(8)))  float  v8f;

union Frag {
    uint4  q[2];
    v16bf  v;
};

__device__ __forceinline__ unsigned short f2bf(float f) {
    unsigned int u = __float_as_uint(f);
    unsigned int r = u + 0x7fffu + ((u >> 16) & 1u);   // round to nearest even
    return (unsigned short)(r >> 16);
}

// A fragment of a 16x32 bf16 tile (M x K), row-major source, rstride in shorts.
// Layout (ISA 7.12.2): lanes 0-15 -> K half 0, lanes 16-31 -> K half 1.
__device__ __forceinline__ v16bf load_a_frag(const unsigned short* base, int rstride,
                                             int m, int kh, int kb) {
    Frag f;
    const unsigned short* p = base + (size_t)m * rstride + kb * 32 + kh * 8;
    f.q[0] = *(const uint4*)(p);        // K = kb*32 + kh*8 + [0..7]   -> VGPR0..3
    f.q[1] = *(const uint4*)(p + 16);   // K = kb*32 + 16 + kh*8 + [0..7] -> VGPR4..7
    return f.v;
}

// B fragment of a 32x16 bf16 tile (K x N) where B[k][n] = src[n][k] (src row-major).
// Layout: same K order as A: lanes 0-15 hold K=0..7,16..23; lanes 16-31 hold K=8..15,24..31 for column n = lane&15.
__device__ __forceinline__ v16bf load_b_frag(const unsigned short* base, int rstride,
                                             int n, int kh, int kb) {
    Frag f;
    const unsigned short* p = base + (size_t)n * rstride + kb * 32 + kh * 8;
    f.q[0] = *(const uint4*)(p);
    f.q[1] = *(const uint4*)(p + 16);
    return f.v;
}

// ---------------------------------------------------------------------------
// Kernel 1: per-row norm; emit bf16 x (xb) and bf16 normalized x (xn), padded.
// ---------------------------------------------------------------------------
__global__ __launch_bounds__(128) void prep_x(const float* __restrict__ x,
                                              unsigned short* __restrict__ xb,
                                              unsigned short* __restrict__ xn) {
    __shared__ float red[128];
    int bn = blockIdx.x;
    int b = bn / N_, n = bn % N_;
    int tid = threadIdx.x;
    const float v = x[((size_t)b * N_ + n) * D_ + tid];
    red[tid] = v * v;
    __syncthreads();
    for (int s = 64; s > 0; s >>= 1) {
        if (tid < s) red[tid] += red[tid + s];
        __syncthreads();
    }
    float inv = 1.0f / fmaxf(sqrtf(red[0]), 1e-12f);
    size_t o = ((size_t)b * NPAD + n) * D_ + tid;
    xb[o] = f2bf(v);
    xn[o] = f2bf(v * inv);
}

// ---------------------------------------------------------------------------
// Kernel 2: fp32 -> bf16 weight conversion.
// ---------------------------------------------------------------------------
__global__ void f2bf_kernel(const float* __restrict__ src,
                            unsigned short* __restrict__ dst, int n) {
    int i = blockIdx.x * blockDim.x + threadIdx.x;
    if (i < n) dst[i] = f2bf(src[i]);
}

// ---------------------------------------------------------------------------
// Kernel 3: sim = xn @ xn^T (per batch), bf16 WMMA, fp32 accumulate.
// One wave per block: 16 queries x 47 key tiles. Key tiles processed in pairs
// with two independent accumulator chains (hides WMMA RAW latency) and all
// B-fragment loads issued before the WMMA block (single partial-wait clause).
// Stores are unconditional into the NPAD-padded sim buffer (no exec-mask
// branching); padded rows/cols are never consumed downstream.
// ---------------------------------------------------------------------------
__global__ __launch_bounds__(32) void sim_wmma(const unsigned short* __restrict__ xn,
                                               float* __restrict__ sim) {
    const int kChunks = 4;
    const int tilesPerChunk = QTILES / kChunks;   // 47 (odd)
    int lin   = blockIdx.x;
    int chunk = lin % kChunks; lin /= kChunks;
    int qt    = lin % QTILES;
    int b     = lin / QTILES;

    int lane = threadIdx.x;
    int m  = lane & 15;      // row of A / col of B & D
    int kh = lane >> 4;

    const unsigned short* xbat = xn + (size_t)b * NPAD * D_;
    int qbase = qt * 16;
    const unsigned short* arow = xbat + (size_t)qbase * D_;

    // A fragments for K = 0..127 (kept live across the key loop).
    v16bf a0 = load_a_frag(arow, D_, m, kh, 0);
    v16bf a1 = load_a_frag(arow, D_, m, kh, 1);
    v16bf a2 = load_a_frag(arow, D_, m, kh, 2);
    v16bf a3 = load_a_frag(arow, D_, m, kh, 3);

    // Destination base: lane writes rows (i + 8*kh), column m of each tile.
    float* dbase = sim + ((size_t)b * NPAD + qbase) * (size_t)NPAD;

    int ktBeg = chunk * tilesPerChunk;
    int ktEnd = ktBeg + tilesPerChunk;
    int kt = ktBeg;

    // Peel one solo tile (tilesPerChunk is odd).
    {
        const unsigned short* krow = xbat + (size_t)kt * 16 * D_;
        v16bf b0 = load_b_frag(krow, D_, m, kh, 0);
        v16bf b1 = load_b_frag(krow, D_, m, kh, 1);
        v16bf b2 = load_b_frag(krow, D_, m, kh, 2);
        v16bf b3 = load_b_frag(krow, D_, m, kh, 3);
        v8f acc = {};
        acc = __builtin_amdgcn_wmma_f32_16x16x32_bf16(false, a0, false, b0, (short)0, acc, false, false);
        acc = __builtin_amdgcn_wmma_f32_16x16x32_bf16(false, a1, false, b1, (short)0, acc, false, false);
        acc = __builtin_amdgcn_wmma_f32_16x16x32_bf16(false, a2, false, b2, (short)0, acc, false, false);
        acc = __builtin_amdgcn_wmma_f32_16x16x32_bf16(false, a3, false, b3, (short)0, acc, false, false);
        float* d0 = dbase + (size_t)kt * 16 + m;
#pragma unroll
        for (int i = 0; i < 8; ++i)
            d0[(size_t)(i + kh * 8) * NPAD] = acc[i];
        ++kt;
    }

    // Paired key tiles: two independent WMMA chains per iteration.
    for (; kt < ktEnd; kt += 2) {
        const unsigned short* krow0 = xbat + (size_t)kt * 16 * D_;
        const unsigned short* krow1 = krow0 + (size_t)16 * D_;
        // Prefetch the next pair's first tile (lowers to global_prefetch_b8).
        if (kt + 2 < ktEnd)
            __builtin_prefetch(krow0 + (size_t)32 * D_ + (size_t)m * D_, 0, 1);

        v16bf b0 = load_b_frag(krow0, D_, m, kh, 0);
        v16bf b1 = load_b_frag(krow0, D_, m, kh, 1);
        v16bf b2 = load_b_frag(krow0, D_, m, kh, 2);
        v16bf b3 = load_b_frag(krow0, D_, m, kh, 3);
        v16bf c0 = load_b_frag(krow1, D_, m, kh, 0);
        v16bf c1 = load_b_frag(krow1, D_, m, kh, 1);
        v16bf c2 = load_b_frag(krow1, D_, m, kh, 2);
        v16bf c3 = load_b_frag(krow1, D_, m, kh, 3);

        v8f acc0 = {};
        v8f acc1 = {};
        acc0 = __builtin_amdgcn_wmma_f32_16x16x32_bf16(false, a0, false, b0, (short)0, acc0, false, false);
        acc1 = __builtin_amdgcn_wmma_f32_16x16x32_bf16(false, a0, false, c0, (short)0, acc1, false, false);
        acc0 = __builtin_amdgcn_wmma_f32_16x16x32_bf16(false, a1, false, b1, (short)0, acc0, false, false);
        acc1 = __builtin_amdgcn_wmma_f32_16x16x32_bf16(false, a1, false, c1, (short)0, acc1, false, false);
        acc0 = __builtin_amdgcn_wmma_f32_16x16x32_bf16(false, a2, false, b2, (short)0, acc0, false, false);
        acc1 = __builtin_amdgcn_wmma_f32_16x16x32_bf16(false, a2, false, c2, (short)0, acc1, false, false);
        acc0 = __builtin_amdgcn_wmma_f32_16x16x32_bf16(false, a3, false, b3, (short)0, acc0, false, false);
        acc1 = __builtin_amdgcn_wmma_f32_16x16x32_bf16(false, a3, false, c3, (short)0, acc1, false, false);

        float* d0 = dbase + (size_t)kt * 16 + m;
        float* d1 = d0 + 16;
#pragma unroll
        for (int i = 0; i < 8; ++i) {
            size_t ro = (size_t)(i + kh * 8) * NPAD;
            d0[ro] = acc0[i];
            d1[ro] = acc1[i];
        }
    }
}

// ---------------------------------------------------------------------------
// Kernel 4: per-query top-40 over valid keys (deterministic argmax rounds,
// ties broken toward the smallest index, matching jax.lax.top_k).
// ---------------------------------------------------------------------------
__global__ __launch_bounds__(128) void topk_kernel(const float* __restrict__ sim,
                                                   const unsigned char* __restrict__ valid,
                                                   int* __restrict__ topk) {
    __shared__ float row[NPAD];
    __shared__ float rmax[128];
    __shared__ int   ridx[128];
    int bq = blockIdx.x;
    int b = bq / N_, q = bq % N_;
    int tid = threadIdx.x;
    int* out = topk + ((size_t)b * N_ + q) * TOPK_;
    const unsigned char* vb = valid + (size_t)b * N_;
    if (!vb[q]) {                        // padded query: unused downstream
        for (int i = tid; i < TOPK_; i += 128) out[i] = -1;
        return;
    }
    const float* srow = sim + ((size_t)b * NPAD + q) * (size_t)NPAD;
    for (int k = tid; k < N_; k += 128) row[k] = vb[k] ? srow[k] : NEGF;
    __syncthreads();
    for (int r = 0; r < TOPK_; ++r) {
        float m = NEGF;
        int mi = N_;
        for (int k = tid; k < N_; k += 128) {
            float v = row[k];
            if (v > m) { m = v; mi = k; }
        }
        rmax[tid] = m; ridx[tid] = mi;
        __syncthreads();
        if (tid == 0) {
            float bm = rmax[0]; int bi = ridx[0];
            for (int t = 1; t < 128; ++t) {
                if (rmax[t] > bm || (rmax[t] == bm && ridx[t] < bi)) { bm = rmax[t]; bi = ridx[t]; }
            }
            out[r] = bi;
            row[bi] = NEGF;
        }
        __syncthreads();
    }
}

// ---------------------------------------------------------------------------
// Kernel 5: generic projection out = inp @ W^T + bias via bf16 WMMA.
// Used for QKV (outDim=384) and output projection (outDim=128).
// ---------------------------------------------------------------------------
__global__ __launch_bounds__(32) void proj_wmma(const unsigned short* __restrict__ inp,
                                                const unsigned short* __restrict__ w,
                                                const float* __restrict__ bias,
                                                float* __restrict__ out, int outDim) {
    int jTiles = outDim >> 4;
    int lin = blockIdx.x;
    int jt = lin % jTiles; lin /= jTiles;
    int qt = lin % QTILES;
    int b  = lin / QTILES;

    int lane = threadIdx.x;
    int m = lane & 15, kh = lane >> 4;
    int qbase = qt * 16, jbase = jt * 16;

    const unsigned short* arow = inp + ((size_t)b * NPAD + qbase) * D_;
    const unsigned short* wrow = w + (size_t)jbase * D_;   // B[k][n] = W[jbase+n][k]

    // Issue all fragment loads before the WMMA block.
    v16bf a0 = load_a_frag(arow, D_, m, kh, 0);
    v16bf a1 = load_a_frag(arow, D_, m, kh, 1);
    v16bf a2 = load_a_frag(arow, D_, m, kh, 2);
    v16bf a3 = load_a_frag(arow, D_, m, kh, 3);
    v16bf b0 = load_b_frag(wrow, D_, m, kh, 0);
    v16bf b1 = load_b_frag(wrow, D_, m, kh, 1);
    v16bf b2 = load_b_frag(wrow, D_, m, kh, 2);
    v16bf b3 = load_b_frag(wrow, D_, m, kh, 3);

    v8f acc = {};
    acc = __builtin_amdgcn_wmma_f32_16x16x32_bf16(false, a0, false, b0, (short)0, acc, false, false);
    acc = __builtin_amdgcn_wmma_f32_16x16x32_bf16(false, a1, false, b1, (short)0, acc, false, false);
    acc = __builtin_amdgcn_wmma_f32_16x16x32_bf16(false, a2, false, b2, (short)0, acc, false, false);
    acc = __builtin_amdgcn_wmma_f32_16x16x32_bf16(false, a3, false, b3, (short)0, acc, false, false);

    float bv = bias[jbase + m];
#pragma unroll
    for (int i = 0; i < 8; ++i) {
        int mm = i + kh * 8;
        out[((size_t)b * NPAD + qbase + mm) * outDim + jbase + m] = acc[i] + bv;
    }
}

// ---------------------------------------------------------------------------
// Kernel 6: sparse attention. One block per (b,q).
// Valid q: keys = (top-40 ∩ valid) ∪ {q}. Invalid q: keys = all valid keys
// (fallback: all keys if none valid). Writes bf16 context.
// ---------------------------------------------------------------------------
__global__ __launch_bounds__(128) void attn_sparse(const float* __restrict__ qkv,
                                                   const int* __restrict__ topk,
                                                   const unsigned char* __restrict__ valid,
                                                   unsigned short* __restrict__ ctxb) {
    __shared__ int   list[NPAD];
    __shared__ float qvec[D_];
    __shared__ float sc[NPAD];
    __shared__ float red[128];
    __shared__ int   s_cnt;

    int bq = blockIdx.x;
    int b = bq / N_, q = bq % N_;
    int tid = threadIdx.x;
    const unsigned char* vb = valid + (size_t)b * N_;
    const float* qrow = qkv + ((size_t)b * NPAD + q) * 384;
    qvec[tid] = qrow[tid];               // q part = first 128

    if (tid == 0) {                      // deterministic list build
        int cnt = 0;
        if (vb[q]) {
            const int* tk = topk + ((size_t)b * N_ + q) * TOPK_;
            bool hasSelf = false;
            for (int i = 0; i < TOPK_; ++i) {
                int idx = tk[i];
                if (idx >= 0 && idx < N_ && vb[idx]) {
                    list[cnt++] = idx;
                    if (idx == q) hasSelf = true;
                }
            }
            if (!hasSelf) list[cnt++] = q;
        } else {
            for (int k = 0; k < N_; ++k) if (vb[k]) list[cnt++] = k;
            if (cnt == 0) { for (int k = 0; k < N_; ++k) list[k] = k; cnt = N_; }
        }
        s_cnt = cnt;
    }
    __syncthreads();
    const int cnt = s_cnt;
    const float scale = 0.17677669529663687f;   // 1/sqrt(32)

    for (int h = 0; h < H_; ++h) {
        const float* qh = qvec + h * HD_;
        for (int i = tid; i < cnt; i += 128) {
            const float* krow = qkv + ((size_t)b * NPAD + list[i]) * 384 + 128 + h * HD_;
            float d = 0.f;
#pragma unroll
            for (int t = 0; t < HD_; ++t) d += qh[t] * krow[t];
            sc[i] = d * scale;
        }
        __syncthreads();
        // max
        float m = NEGF;
        for (int i = tid; i < cnt; i += 128) m = fmaxf(m, sc[i]);
        red[tid] = m; __syncthreads();
        for (int s = 64; s > 0; s >>= 1) { if (tid < s) red[tid] = fmaxf(red[tid], red[tid + s]); __syncthreads(); }
        float mx = red[0]; __syncthreads();
        // exp + sum
        float sum = 0.f;
        for (int i = tid; i < cnt; i += 128) { float e = __expf(sc[i] - mx); sc[i] = e; sum += e; }
        red[tid] = sum; __syncthreads();
        for (int s = 64; s > 0; s >>= 1) { if (tid < s) red[tid] += red[tid + s]; __syncthreads(); }
        float inv = 1.0f / red[0]; __syncthreads();
        // ctx: thread = (seg, d)
        int d = tid & 31, seg = tid >> 5;
        float acc = 0.f;
        for (int i = seg; i < cnt; i += 4)
            acc += sc[i] * qkv[((size_t)b * NPAD + list[i]) * 384 + 256 + h * HD_ + d];
        red[tid] = acc; __syncthreads();
        if (tid < 32) {
            float c = (red[tid] + red[tid + 32] + red[tid + 64] + red[tid + 96]) * inv;
            ctxb[((size_t)b * NPAD + q) * D_ + h * HD_ + tid] = f2bf(c);
        }
        __syncthreads();
    }
}

// ---------------------------------------------------------------------------
// Kernel 7: residual + LayerNorm.
// ---------------------------------------------------------------------------
__global__ __launch_bounds__(128) void ln_kernel(const float* __restrict__ x,
                                                 const float* __restrict__ att,
                                                 const float* __restrict__ gamma,
                                                 const float* __restrict__ beta,
                                                 float* __restrict__ out) {
    __shared__ float red[128];
    int bn = blockIdx.x;
    int b = bn / N_, n = bn % N_;
    int tid = threadIdx.x;
    float y = x[((size_t)b * N_ + n) * D_ + tid] + att[((size_t)b * NPAD + n) * D_ + tid];
    red[tid] = y; __syncthreads();
    for (int s = 64; s > 0; s >>= 1) { if (tid < s) red[tid] += red[tid + s]; __syncthreads(); }
    float mu = red[0] * (1.0f / 128.0f); __syncthreads();
    float dv = y - mu;
    red[tid] = dv * dv; __syncthreads();
    for (int s = 64; s > 0; s >>= 1) { if (tid < s) red[tid] += red[tid + s]; __syncthreads(); }
    float var = red[0] * (1.0f / 128.0f);
    out[((size_t)b * N_ + n) * D_ + tid] = dv * rsqrtf(var + 1e-5f) * gamma[tid] + beta[tid];
}

// ---------------------------------------------------------------------------
extern "C" void kernel_launch(void* const* d_in, const int* in_sizes, int n_in,
                              void* d_out, int out_size, void* d_ws, size_t ws_size,
                              hipStream_t stream) {
    const float*         x     = (const float*)d_in[0];
    const unsigned char* valid = (const unsigned char*)d_in[1];  // jnp bool = 1 byte
    const float*         wqkvf = (const float*)d_in[2];          // (384,128)
    const float*         bqkv  = (const float*)d_in[3];          // (384,)
    const float*         woutf = (const float*)d_in[4];          // (128,128)
    const float*         bout  = (const float*)d_in[5];          // (128,)
    const float*         gamma = (const float*)d_in[6];
    const float*         beta  = (const float*)d_in[7];
    float*               out   = (float*)d_out;

    char* ws = (char*)d_ws;
    auto al = [](size_t v) { return (v + 255) & ~(size_t)255; };
    size_t o = 0;
    unsigned short* xb   = (unsigned short*)(ws + o); o = al(o + (size_t)B_ * NPAD * D_ * 2);
    unsigned short* xn   = (unsigned short*)(ws + o); o = al(o + (size_t)B_ * NPAD * D_ * 2);
    unsigned short* wqkv = (unsigned short*)(ws + o); o = al(o + (size_t)384 * D_ * 2);
    unsigned short* wout = (unsigned short*)(ws + o); o = al(o + (size_t)D_ * D_ * 2);
    float*          qkv  = (float*)(ws + o);          o = al(o + (size_t)B_ * NPAD * 384 * 4);
    unsigned short* ctxb = (unsigned short*)(ws + o); o = al(o + (size_t)B_ * NPAD * D_ * 2);
    float*          att  = (float*)(ws + o);          o = al(o + (size_t)B_ * NPAD * D_ * 4);
    int*            tpk  = (int*)(ws + o);            o = al(o + (size_t)B_ * N_ * TOPK_ * 4);
    float*          sim  = (float*)(ws + o);          o = al(o + (size_t)B_ * NPAD * NPAD * 4);
    (void)o; (void)ws_size; (void)in_sizes; (void)n_in; (void)out_size;

    // 1) normalize + bf16 convert
    prep_x<<<B_ * N_, 128, 0, stream>>>(x, xb, xn);
    // 2) weight conversion
    f2bf_kernel<<<(384 * D_ + 255) / 256, 256, 0, stream>>>(wqkvf, wqkv, 384 * D_);
    f2bf_kernel<<<(D_ * D_ + 255) / 256, 256, 0, stream>>>(woutf, wout, D_ * D_);
    // 3) similarity GEMM (WMMA bf16)
    sim_wmma<<<B_ * QTILES * 4, 32, 0, stream>>>(xn, sim);
    // 4) top-40 per valid query
    topk_kernel<<<B_ * N_, 128, 0, stream>>>(sim, valid, tpk);
    // 5) QKV projection (WMMA bf16)
    proj_wmma<<<B_ * QTILES * (384 / 16), 32, 0, stream>>>(xb, wqkv, bqkv, qkv, 384);
    // 6) sparse attention
    attn_sparse<<<B_ * N_, 128, 0, stream>>>(qkv, tpk, valid, ctxb);
    // 7) output projection (WMMA bf16)
    proj_wmma<<<B_ * QTILES * (D_ / 16), 32, 0, stream>>>(ctxb, wout, bout, att, D_);
    // 8) residual + LayerNorm
    ln_kernel<<<B_ * N_, 128, 0, stream>>>(x, att, gamma, beta, out);
}
